// SmaacActor_78735340470663
// MI455X (gfx1250) — compile-verified, matrix operand order checked by reading, not run on hardware
//
#include <hip/hip_runtime.h>
#include <hip/hip_bf16.h>
#include <math.h>

#define NODE 1024
#define DM   128
#define NB   16
#define NH   4
#define DKH  32
#define ACTD 128
#define SROW 1032   // padded LDS row stride (floats); keeps 16B alignment of 8-float runs

typedef __attribute__((ext_vector_type(16))) _Float16 v16h;
typedef __attribute__((ext_vector_type(8)))  float    v8f;

// 16-bit A/B fragment K-offset within a 32-wide K chunk (ISA 7.12.2 table):
// lanes 0-15: v0..3 -> K=0..7, v4..7 -> K=16..23 ; lanes 16-31: +8.
__device__ __forceinline__ int frag_k(int hi, int e) {
    return ((e < 8) ? 0 : 16) + hi + (e & 7);
}

__device__ __forceinline__ v16h cvt_frag(float4 a, float4 b, float4 c, float4 d) {
    v16h f;
    f[0]  = (_Float16)a.x; f[1]  = (_Float16)a.y; f[2]  = (_Float16)a.z; f[3]  = (_Float16)a.w;
    f[4]  = (_Float16)b.x; f[5]  = (_Float16)b.y; f[6]  = (_Float16)b.z; f[7]  = (_Float16)b.w;
    f[8]  = (_Float16)c.x; f[9]  = (_Float16)c.y; f[10] = (_Float16)c.z; f[11] = (_Float16)c.w;
    f[12] = (_Float16)d.x; f[13] = (_Float16)d.y; f[14] = (_Float16)d.z; f[15] = (_Float16)d.w;
    return f;
}

// Fragment whose two 8-element halves are contiguous in memory:
// elements 0..7 <- p[hi .. hi+7], elements 8..15 <- p[16+hi .. 23+hi].
// p + hi is 16B-aligned at every call site -> lowers to b128 loads.
__device__ __forceinline__ v16h load_frag_contig(const float* __restrict__ p, int hi) {
    const float4* q0 = reinterpret_cast<const float4*>(p + hi);
    const float4* q1 = reinterpret_cast<const float4*>(p + 16 + hi);
    return cvt_frag(q0[0], q0[1], q1[0], q1[1]);
}
__device__ __forceinline__ v16h load_frag_contig_lds(const float* p, int hi) {
    const float4* q0 = reinterpret_cast<const float4*>(p + hi);
    const float4* q1 = reinterpret_cast<const float4*>(p + 16 + hi);
    return cvt_frag(q0[0], q0[1], q1[0], q1[1]);
}

__device__ __forceinline__ float wred_sum(float s) {
    for (int m = 16; m; m >>= 1) s += __shfl_xor(s, m, 32);
    return s;
}
__device__ __forceinline__ float wred_max(float s) {
    for (int m = 16; m; m >>= 1) s = fmaxf(s, __shfl_xor(s, m, 32));
    return s;
}

// ---------------------------------------------------------------------------
// WMMA GEMM: C[M,N] = act(A[M,K] @ W[K,N] + bias)
// One 16x64 output strip per wave: shared A fragment, 4 accumulators,
// 4 back-to-back v_wmma_f32_16x16x32_f16 per K chunk. N % 64 == 0.
// act: 0 = none, 1 = relu, 2 = clip(-10, 1)
// ---------------------------------------------------------------------------
__global__ void gemm_wmma_kernel(const float* __restrict__ A,
                                 const float* __restrict__ W,
                                 const float* __restrict__ bias,
                                 float* __restrict__ C,
                                 int M, int N, int K, int act)
{
    const int lane = threadIdx.x & 31;
    const int wave = blockIdx.x * (blockDim.x >> 5) + (threadIdx.x >> 5);
    const int ngrp  = N >> 6;                  // 4 n-tiles per wave
    const int tiles = (M >> 4) * ngrp;
    if (wave >= tiles) return;

    const int mt = wave / ngrp;
    const int ng = wave % ngrp;
    const int l15 = lane & 15;
    const int hi  = (lane & 16) ? 8 : 0;
    const int row  = (mt << 4) + l15;          // A-fragment row for this lane
    const int col0 = (ng << 6) + l15;          // first B-fragment column

    const float* __restrict__ arow = A + (size_t)row * K;
    v8f acc0 = {}, acc1 = {}, acc2 = {}, acc3 = {};

    for (int kc = 0; kc < K; kc += 32) {
        const v16h af = load_frag_contig(arow + kc, hi);
        v16h b0, b1, b2, b3;
#pragma unroll
        for (int e = 0; e < 16; ++e) {
            const float* __restrict__ wr = W + (size_t)(kc + frag_k(hi, e)) * N + col0;
            b0[e] = (_Float16)wr[0];
            b1[e] = (_Float16)wr[16];
            b2[e] = (_Float16)wr[32];
            b3[e] = (_Float16)wr[48];
        }
        acc0 = __builtin_amdgcn_wmma_f32_16x16x32_f16(false, af, false, b0, (short)0, acc0, false, false);
        acc1 = __builtin_amdgcn_wmma_f32_16x16x32_f16(false, af, false, b1, (short)0, acc1, false, false);
        acc2 = __builtin_amdgcn_wmma_f32_16x16x32_f16(false, af, false, b2, (short)0, acc2, false, false);
        acc3 = __builtin_amdgcn_wmma_f32_16x16x32_f16(false, af, false, b3, (short)0, acc3, false, false);
    }

    v8f* accs[4] = { &acc0, &acc1, &acc2, &acc3 };
#pragma unroll
    for (int tI = 0; tI < 4; ++tI) {
        const int col = col0 + 16 * tI;
        const float bv = bias ? bias[col] : 0.f;
#pragma unroll
        for (int j = 0; j < 8; ++j) {
            const int r = (mt << 4) + j + hi;  // C layout: vgpr j -> M = j (+8 hi half)
            float v = (*accs[tI])[j] + bv;
            if (act == 1)      v = v > 0.f ? v : 0.f;
            else if (act == 2) v = fminf(fmaxf(v, -10.f), 1.f);
            C[(size_t)r * N + col] = v;
        }
    }
}

// ---------------------------------------------------------------------------
// Fused masked attention: per (batch, head, 16-query tile), one wave.
// scores(16x1024) -> LDS, mask+scale, softmax, P@V -> out[b, q, h*32+d]
// ---------------------------------------------------------------------------
__global__ void attn_kernel(const float* __restrict__ q,
                            const float* __restrict__ k,
                            const float* __restrict__ v,
                            const int*   __restrict__ adj,
                            float* __restrict__ out)
{
    extern __shared__ float s[];               // [16][SROW]
    const int lane = threadIdx.x & 31;
    const int qt = blockIdx.x;                 // 0..63
    const int h  = blockIdx.y;                 // 0..3
    const int b  = blockIdx.z;                 // 0..15
    const int l15 = lane & 15;
    const int hi  = (lane & 16) ? 8 : 0;
    const float scale = 0.17677669529663687f;  // 1/sqrt(32)

    // Q fragment (A, 16x32): row = qt*16 + l15, cols = head's DK slice. Loaded once.
    const int qrow = (qt << 4) + l15;
    const v16h aq = load_frag_contig(q + ((size_t)b * NODE + qrow) * DM + h * DKH, hi);

    // ---- Phase 1: scores = (Q @ K^T) * scale, adjacency-masked, into LDS ----
    for (int kt = 0; kt < 64; ++kt) {
        // B fragment (32x16): B[d][n] = K[kt*16+n][h*32+d]; contiguous along d.
        const int krow = (kt << 4) + l15;
        const v16h bk = load_frag_contig(k + ((size_t)b * NODE + krow) * DM + h * DKH, hi);

        v8f c = {};
        c = __builtin_amdgcn_wmma_f32_16x16x32_f16(false, aq, false, bk, (short)0, c, false, false);

#pragma unroll
        for (int j = 0; j < 8; ++j) {
            const int m  = j + hi;
            const int kk = (kt << 4) + l15;
            const int a  = adj[((size_t)b * NODE + ((qt << 4) + m)) * NODE + kk];
            s[m * SROW + kk] = a ? c[j] * scale : -1.0e9f;
        }
    }

    // ---- Phase 2: row-wise softmax in LDS; keep 1/sum in registers ----
    float inv8[8];
#pragma unroll
    for (int r = 0; r < 16; ++r) {
        float mx = -3.0e38f;
        for (int i = 0; i < 32; ++i)
            mx = fmaxf(mx, s[r * SROW + lane + 32 * i]);
        mx = wred_max(mx);
        float sum = 0.f;
        for (int i = 0; i < 32; ++i) {
            const int idx = r * SROW + lane + 32 * i;
            const float e = __expf(s[idx] - mx);
            s[idx] = e;
            sum += e;
        }
        sum = wred_sum(sum);
        const float iv = 1.f / sum;
        if (r < 8) { if (hi == 0) inv8[r]     = iv; }
        else       { if (hi == 8) inv8[r - 8] = iv; }
    }

    // ---- Phase 3: out = P(16x1024) @ V(1024x32), two 16-wide N halves ----
    v8f a0 = {}, a1 = {};
    for (int kc = 0; kc < 32; ++kc) {
        const int m = l15;
        const v16h ap = load_frag_contig_lds(&s[m * SROW + kc * 32], hi);
        v16h b0, b1;
#pragma unroll
        for (int e = 0; e < 16; ++e) {
            const int kk = kc * 32 + frag_k(hi, e);
            const float* __restrict__ vp = v + ((size_t)b * NODE + kk) * DM + h * DKH;
            b0[e] = (_Float16)vp[l15];
            b1[e] = (_Float16)vp[16 + l15];
        }
        a0 = __builtin_amdgcn_wmma_f32_16x16x32_f16(false, ap, false, b0, (short)0, a0, false, false);
        a1 = __builtin_amdgcn_wmma_f32_16x16x32_f16(false, ap, false, b1, (short)0, a1, false, false);
    }

#pragma unroll
    for (int j = 0; j < 8; ++j) {
        const int m   = j + hi;
        const int row = (qt << 4) + m;
        const float is = inv8[j];
        float* __restrict__ op = out + ((size_t)b * NODE + row) * DM + h * DKH;
        op[l15]      = a0[j] * is;
        op[16 + l15] = a1[j] * is;
    }
}

// ---------------------------------------------------------------------------
// LayerNorm with residual: out = LN(a + res) * g + b   (rows of DM=128)
// ---------------------------------------------------------------------------
__global__ void ln_res_kernel(const float* __restrict__ a,
                              const float* __restrict__ res,
                              const float* __restrict__ g,
                              const float* __restrict__ bb,
                              float* __restrict__ out, int rows)
{
    const int lane = threadIdx.x & 31;
    const int row  = blockIdx.x * (blockDim.x >> 5) + (threadIdx.x >> 5);
    if (row >= rows) return;
    const float* __restrict__ pa = a   + (size_t)row * DM;
    const float* __restrict__ pr = res + (size_t)row * DM;
    float x[4];
    float sum = 0.f;
#pragma unroll
    for (int i = 0; i < 4; ++i) {
        const int c = lane + 32 * i;
        x[i] = pa[c] + pr[c];
        sum += x[i];
    }
    const float mu = wred_sum(sum) * (1.f / DM);
    float vs = 0.f;
#pragma unroll
    for (int i = 0; i < 4; ++i) { const float d = x[i] - mu; vs += d * d; }
    const float inv = rsqrtf(wred_sum(vs) * (1.f / DM) + 1e-6f);
    float* __restrict__ po = out + (size_t)row * DM;
#pragma unroll
    for (int i = 0; i < 4; ++i) {
        const int c = lane + 32 * i;
        po[c] = (x[i] - mu) * inv * g[c] + bb[c];
    }
}

// ---------------------------------------------------------------------------
// down = h @ down_w + down_b  ->  state[b, n] and z = leaky_relu(state)
// ---------------------------------------------------------------------------
__global__ void down_kernel(const float* __restrict__ h,
                            const float* __restrict__ w,
                            const float* __restrict__ bias,
                            float* __restrict__ state,
                            float* __restrict__ z)
{
    const int lane = threadIdx.x & 31;
    const int row  = blockIdx.x * (blockDim.x >> 5) + (threadIdx.x >> 5);
    const float* __restrict__ p = h + (size_t)row * DM;
    float s = 0.f;
#pragma unroll
    for (int i = 0; i < 4; ++i) { const int c = lane + 32 * i; s += p[c] * w[c]; }
    s = wred_sum(s);
    if (lane == 0) {
        const float val = s + bias[0];
        const int bb = row / NODE, n = row % NODE;
        const size_t o = (size_t)bb * (2 * NODE) + n;
        state[o] = val;
        z[o] = val > 0.f ? val : 0.01f * val;
    }
}

__global__ void tpart_kernel(const float* __restrict__ t,
                             float* __restrict__ state,
                             float* __restrict__ z)
{
    const int i = blockIdx.x * blockDim.x + threadIdx.x;   // B*NODE
    const int bb = i / NODE, n = i % NODE;
    const float val = t[i];
    const size_t o = (size_t)bb * (2 * NODE) + NODE + n;
    state[o] = val;
    z[o] = val > 0.f ? val : 0.01f * val;
}

// ---------------------------------------------------------------------------
extern "C" void kernel_launch(void* const* d_in, const int* in_sizes, int n_in,
                              void* d_out, int out_size, void* d_ws, size_t ws_size,
                              hipStream_t stream)
{
    (void)in_sizes; (void)n_in; (void)out_size; (void)ws_size;

    const float* x   = (const float*)d_in[0];
    const float* t   = (const float*)d_in[1];
    const int*   adj = (const int*)d_in[2];

    float* ws = (float*)d_ws;
    const size_t SZ = (size_t)NB * NODE * DM;
    float* h1  = ws;
    float* h2  = ws + 1 * SZ;
    float* q   = ws + 2 * SZ;
    float* kb  = ws + 3 * SZ;
    float* vb  = ws + 4 * SZ;
    float* ao  = ws + 5 * SZ;
    float* tmp = ws + 6 * SZ;
    float* z   = ws + 7 * SZ;            // [16, 2048]

    float* outp  = (float*)d_out;
    float* mu    = outp;                 // [16,128]
    float* ls    = outp + NB * ACTD;     // [16,128]
    float* state = outp + 2 * NB * ACTD; // [16,2048]

    auto gemm = [&](const float* A, const float* W, const float* bias,
                    float* C, int M, int N, int K, int act) {
        const int tiles  = (M / 16) * (N / 64);
        const int blocks = (tiles + 3) / 4;
        gemm_wmma_kernel<<<blocks, 128, 0, stream>>>(A, W, bias, C, M, N, K, act);
    };

    const int MROWS = NB * NODE;                       // 16384
    const size_t attn_lds = (size_t)16 * SROW * sizeof(float);

    auto layer = [&](const float* in, float* lout, int p) {
        const float* wq  = (const float*)d_in[p + 0];
        const float* wk  = (const float*)d_in[p + 1];
        const float* wv  = (const float*)d_in[p + 2];
        const float* fc  = (const float*)d_in[p + 3];
        const float* l1g = (const float*)d_in[p + 4];
        const float* l1b = (const float*)d_in[p + 5];
        const float* w1  = (const float*)d_in[p + 6];
        const float* b1  = (const float*)d_in[p + 7];
        const float* w2  = (const float*)d_in[p + 8];
        const float* b2  = (const float*)d_in[p + 9];
        const float* l2g = (const float*)d_in[p + 10];
        const float* l2b = (const float*)d_in[p + 11];

        gemm(in, wq, nullptr, q,  MROWS, DM, DM, 0);
        gemm(in, wk, nullptr, kb, MROWS, DM, DM, 0);
        gemm(in, wv, nullptr, vb, MROWS, DM, DM, 0);
        attn_kernel<<<dim3(NODE / 16, NH, NB), 32, attn_lds, stream>>>(q, kb, vb, adj, ao);
        gemm(ao, fc, nullptr, tmp, MROWS, DM, DM, 0);
        ln_res_kernel<<<MROWS / 4, 128, 0, stream>>>(tmp, in, l1g, l1b, q, MROWS);  // q := x1
        gemm(q, w1, b1, tmp, MROWS, DM, DM, 1);                                     // relu
        gemm(tmp, w2, b2, kb, MROWS, DM, DM, 0);                                    // kb := ffn out
        ln_res_kernel<<<MROWS / 4, 128, 0, stream>>>(kb, q, l2g, l2b, lout, MROWS);
    };

    layer(x,  h1, 3);
    layer(h1, h2, 15);
    layer(h2, h1, 27);

    down_kernel<<<MROWS / 4, 128, 0, stream>>>(
        h1, (const float*)d_in[39], (const float*)d_in[40], state, z);
    tpart_kernel<<<(NB * NODE) / 256, 256, 0, stream>>>(t, state, z);

    gemm(z, (const float*)d_in[41], (const float*)d_in[42], mu, NB, ACTD, 2 * NODE, 0);
    gemm(z, (const float*)d_in[43], (const float*)d_in[44], ls, NB, ACTD, 2 * NODE, 2);
}
